// MultiHeadAttention_59768764891853
// MI455X (gfx1250) — compile-verified, matrix operand order checked by reading, not run on hardware
//
#include <hip/hip_runtime.h>

// ---------------------------------------------------------------------------
// MI455X (gfx1250) multi-head attention forward.
// bf16 WMMA (v_wmma_f32_16x16x32_bf16) for all matmuls, flash-style fused
// attention (no S x S in HBM: would be >1 GB traffic vs ~100 MB fused at
// 23.3 TB/s), async LDS copy pipeline where the toolchain exposes it,
// DPP 16-lane reductions for the online softmax.
// ---------------------------------------------------------------------------

typedef __attribute__((ext_vector_type(16))) __bf16 v16bf;
typedef __attribute__((ext_vector_type(8)))  __bf16 v8bf;
typedef __attribute__((ext_vector_type(8)))  float  v8f;
typedef __attribute__((ext_vector_type(4)))  int    v4i_;

#define B_DIM 2
#define S_DIM 2048
#define E_DIM 1024
#define H_DIM 16
#define D_DIM 64
#define M_DIM (B_DIM * S_DIM)          // 4096
// scale = D^-0.5 folded with log2(e) so softmax uses exp2 (v_exp_f32)
#define QSCALE 0.18033688011112042f    // 0.125 * 1.4426950408889634

#if defined(__has_builtin)
#if __has_builtin(__builtin_amdgcn_global_load_async_to_lds_b128)
#define HAVE_ASYNC_LDS 1
#endif
#if __has_builtin(__builtin_amdgcn_s_wait_asynccnt)
#define HAVE_WAIT_ASYNC 1
#endif
#if __has_builtin(__builtin_amdgcn_mov_dpp)
#define HAVE_DPP 1
#endif
#endif

__device__ __forceinline__ v8f wmma_bf16(v16bf a, v16bf b, v8f c) {
    // 8 args: (neg_a, A, neg_b, B, c_mod, C, reuse_a, reuse_b)
    return __builtin_amdgcn_wmma_f32_16x16x32_bf16(
        false, a, false, b, (short)0, c, false, false);
}

// Load a 16-element bf16 fragment as two 16B chunks (16B-aligned LDS/global).
__device__ __forceinline__ v16bf ldfrag(const __bf16* p) {
    union { v16bf v; v8bf h[2]; } u;
    u.h[0] = *(const v8bf*)(p);
    u.h[1] = *(const v8bf*)(p + 8);
    return u.v;
}

#ifdef HAVE_ASYNC_LDS
// 16B global -> LDS async copy (ASYNCcnt-tracked, no VGPR data round-trip).
// Builtin signature (from hipcc diagnostic): (v4i AS1*, v4i AS3*, imm, imm).
__device__ __forceinline__ void async_cp128(const __bf16* g, __bf16* l) {
    __builtin_amdgcn_global_load_async_to_lds_b128(
        (__attribute__((address_space(1))) v4i_*)g,
        (__attribute__((address_space(3))) v4i_*)l, 0, 0);
}
#endif

__device__ __forceinline__ void wait_async0() {
#ifdef HAVE_WAIT_ASYNC
    __builtin_amdgcn_s_wait_asynccnt(0);
#else
    asm volatile("s_wait_asynccnt 0x0" ::: "memory");
#endif
}

// ---- 16-lane (half-wave) reductions: DPP if available, else shfl_xor ------
#ifdef HAVE_DPP
template <int CTRL>
__device__ __forceinline__ float dpp_f32(float x) {
    union { float f; int i; } u, v;
    u.f = x;
    v.i = __builtin_amdgcn_mov_dpp(u.i, CTRL, 0xF, 0xF, true);
    return v.f;
}
__device__ __forceinline__ float red16_max(float x) {
    x = fmaxf(x, dpp_f32<0xB1>(x));   // quad_perm [1,0,3,2]  (xor 1)
    x = fmaxf(x, dpp_f32<0x4E>(x));   // quad_perm [2,3,0,1]  (xor 2)
    x = fmaxf(x, dpp_f32<0x124>(x));  // row_ror:4
    x = fmaxf(x, dpp_f32<0x128>(x));  // row_ror:8
    return x;
}
__device__ __forceinline__ float red16_add(float x) {
    x += dpp_f32<0xB1>(x);
    x += dpp_f32<0x4E>(x);
    x += dpp_f32<0x124>(x);
    x += dpp_f32<0x128>(x);
    return x;
}
#else
__device__ __forceinline__ float red16_max(float x) {
#pragma unroll
    for (int off = 8; off; off >>= 1) x = fmaxf(x, __shfl_xor(x, off, 32));
    return x;
}
__device__ __forceinline__ float red16_add(float x) {
#pragma unroll
    for (int off = 8; off; off >>= 1) x += __shfl_xor(x, off, 32);
    return x;
}
#endif

// ---------------------------------------------------------------------------
// f32 -> bf16 convert (vectorized x4), grid-stride
// ---------------------------------------------------------------------------
__global__ void mha_cvt_bf16(const float4* __restrict__ src,
                             ushort4* __restrict__ dst, int n4) {
    int i = blockIdx.x * blockDim.x + threadIdx.x;
    int stride = gridDim.x * blockDim.x;
    for (; i < n4; i += stride) {
        float4 f = src[i];
        union { __bf16 b[4]; ushort4 u; } o;
        o.b[0] = (__bf16)f.x; o.b[1] = (__bf16)f.y;
        o.b[2] = (__bf16)f.z; o.b[3] = (__bf16)f.w;
        dst[i] = o.u;
    }
}

// ---------------------------------------------------------------------------
// Tiled bf16 WMMA GEMM:  C[M,N] = A[M,K] * Bw[N,K]^T (+ bias)
// 256 threads = 8 waves (4 along M x 2 along N), block tile 128x128,
// wave tile 32x64 (2x4 accumulators), K staged 32 at a time through LDS.
// Async path: double-buffered LDS, async copies for tile k+1 overlap the
// WMMAs on tile k; one barrier per K-step. Fallback: sync staging.
// MODE 0: QKV epilogue -> scatter bf16 into Q/K/V [B,H,S,D], Q pre-scaled.
// MODE 1: out-proj epilogue -> f32 + bias into d_out.
// ---------------------------------------------------------------------------
template <int MODE>
__global__ __launch_bounds__(256) void mha_gemm_bf16(
    const __bf16* __restrict__ A, const __bf16* __restrict__ Bw,
    const float* __restrict__ bias,
    __bf16* __restrict__ qo, __bf16* __restrict__ ko, __bf16* __restrict__ vo,
    float* __restrict__ out, int K) {
#ifdef HAVE_ASYNC_LDS
    __shared__ __bf16 As[2][128 * 40];   // rows padded to 40 (80B, 16B-aligned)
    __shared__ __bf16 Bs[2][128 * 40];
#else
    __shared__ __bf16 As[1][128 * 40];
    __shared__ __bf16 Bs[1][128 * 40];
#endif

    const int tid = threadIdx.x;
    const int lane = tid & 31, wave = tid >> 5;
    const int lane15 = lane & 15, hi = lane >> 4;   // A/B frag K-half select
    const int wm = wave & 3, wn = wave >> 2;
    const int blockM = blockIdx.y * 128, blockN = blockIdx.x * 128;

    v8f acc[2][4] = {};

    const int srow = tid >> 1, shalf = tid & 1;     // 256 thr stage 128x32 bf16
    const __bf16* ag = A  + (size_t)(blockM + srow) * K + shalf * 16;
    const __bf16* bg = Bw + (size_t)(blockN + srow) * K + shalf * 16;
    const int soff = srow * 40 + shalf * 16;

#ifdef HAVE_ASYNC_LDS
    // prologue: async-stage k0 = 0 into buffer 0
    async_cp128(ag, &As[0][soff]);
    async_cp128(ag + 8, &As[0][soff + 8]);
    async_cp128(bg, &Bs[0][soff]);
    async_cp128(bg + 8, &Bs[0][soff + 8]);
#endif

    for (int k0 = 0; k0 < K; k0 += 32) {
#ifdef HAVE_ASYNC_LDS
        const int buf = (k0 >> 5) & 1;
        wait_async0();            // my copies into buf are done
        __syncthreads();          // everyone's copies done; buf^1 free again
        if (k0 + 32 < K) {        // overlap next-tile copy with this compute
            async_cp128(ag + k0 + 32, &As[buf ^ 1][soff]);
            async_cp128(ag + k0 + 40, &As[buf ^ 1][soff + 8]);
            async_cp128(bg + k0 + 32, &Bs[buf ^ 1][soff]);
            async_cp128(bg + k0 + 40, &Bs[buf ^ 1][soff + 8]);
        }
#else
        const int buf = 0;
        __syncthreads();
        *(v8bf*)&As[0][soff]     = *(const v8bf*)(ag + k0);
        *(v8bf*)&As[0][soff + 8] = *(const v8bf*)(ag + k0 + 8);
        *(v8bf*)&Bs[0][soff]     = *(const v8bf*)(bg + k0);
        *(v8bf*)&Bs[0][soff + 8] = *(const v8bf*)(bg + k0 + 8);
        __syncthreads();
#endif
        if (k0 + 128 < K) {       // pull tiles ahead into L2 (global_prefetch)
            __builtin_prefetch(ag + k0 + 128, 0, 1);
            __builtin_prefetch(bg + k0 + 128, 0, 1);
        }

        v16bf af[2], bf[4];
#pragma unroll
        for (int i = 0; i < 2; ++i)
            af[i] = ldfrag(&As[buf][(wm * 32 + i * 16 + lane15) * 40 + hi * 16]);
#pragma unroll
        for (int j = 0; j < 4; ++j)
            bf[j] = ldfrag(&Bs[buf][(wn * 64 + j * 16 + lane15) * 40 + hi * 16]);
#pragma unroll
        for (int i = 0; i < 2; ++i)
#pragma unroll
            for (int j = 0; j < 4; ++j)
                acc[i][j] = wmma_bf16(af[i], bf[j], acc[i][j]);
    }

    // C/D layout: element (m,n) of a 16x16 tile lives in lane n + 16*(m>=8),
    // VGPR m&7.  Per lane: 8 rows (r + 8*hi), one column (lane15).
    if (MODE == 0) {
#pragma unroll
        for (int j = 0; j < 4; ++j) {
            const int ng = blockN + wn * 64 + j * 16 + lane15;  // [0, 3072)
            const float bs = bias[ng];
            const int sel = ng >> 10;          // 0=q 1=k 2=v
            const int e = ng & 1023;
            const int h = e >> 6, dc = e & 63;
            __bf16* dst = (sel == 0) ? qo : ((sel == 1) ? ko : vo);
            const float mul = (sel == 0) ? QSCALE : 1.0f;
#pragma unroll
            for (int i = 0; i < 2; ++i)
#pragma unroll
                for (int r = 0; r < 8; ++r) {
                    const int m = blockM + wm * 32 + i * 16 + r + hi * 8;
                    const int bb = m >> 11, s = m & 2047;
                    dst[(((size_t)(bb * H_DIM + h) * S_DIM + s)) * D_DIM + dc] =
                        (__bf16)((acc[i][j][r] + bs) * mul);
                }
        }
    } else {
#pragma unroll
        for (int j = 0; j < 4; ++j) {
            const int ng = blockN + wn * 64 + j * 16 + lane15;  // [0, 1024)
            const float bs = bias[ng];
#pragma unroll
            for (int i = 0; i < 2; ++i)
#pragma unroll
                for (int r = 0; r < 8; ++r) {
                    const int m = blockM + wm * 32 + i * 16 + r + hi * 8;
                    out[(size_t)m * E_DIM + ng] = acc[i][j][r] + bs;
                }
        }
    }
}

// ---------------------------------------------------------------------------
// Flash attention: one block per (b, h, 64-row q-block), 4 waves / 128 thr.
// Each wave owns 16 q rows: O accum = 16x64 (4x v8f), online softmax stats
// per row kept in the half-wave that owns the row (C-layout compatible).
// Key blocks of 32: K staged row-major (async copy when available), V staged
// TRANSPOSED so the PV B-fragment is a contiguous 32B read. P goes through
// wave-private LDS to convert C-layout -> A-layout for the PV WMMA.
// ---------------------------------------------------------------------------
__global__ __launch_bounds__(128) void mha_flash_attn(
    const __bf16* __restrict__ Qb, const __bf16* __restrict__ Kb,
    const __bf16* __restrict__ Vb, __bf16* __restrict__ Ctx) {
    __shared__ __bf16 Ks[32 * 72];      // [key][d], padded
    __shared__ __bf16 Vt[64 * 40];      // [d][key], padded (transposed V)
    __shared__ __bf16 Pb[4][16 * 40];   // wave-private P tiles

    const int tid = threadIdx.x;
    const int lane = tid & 31, wave = tid >> 5;
    const int lane15 = lane & 15, hi = lane >> 4;
    const int qblk = blockIdx.x, h = blockIdx.y, b = blockIdx.z;
    const size_t bh = (size_t)(b * H_DIM + h);

    const __bf16* Qp = Qb + (bh * S_DIM + qblk * 64 + wave * 16) * D_DIM;
    const __bf16* Kp = Kb + bh * S_DIM * D_DIM;
    const __bf16* Vp = Vb + bh * S_DIM * D_DIM;

    // Q fragments for this wave's 16 rows (2 d-chunks of 32), kept resident.
    v16bf aq[2];
#pragma unroll
    for (int c = 0; c < 2; ++c)
        aq[c] = ldfrag(Qp + (size_t)lane15 * D_DIM + c * 32 + hi * 16);

    v8f acc[4] = {};
    float mi[8], li[8];
#pragma unroll
    for (int r = 0; r < 8; ++r) { mi[r] = -3.0e38f; li[r] = 0.0f; }

    const int srow = tid >> 2, schunk = tid & 3;    // stage 32x64 bf16
    for (int j = 0; j < S_DIM / 32; ++j) {
        __syncthreads();
        const __bf16* ks = Kp + (size_t)(j * 32 + srow) * D_DIM + schunk * 16;
        __bf16* kdst = &Ks[srow * 72 + schunk * 16];
#ifdef HAVE_ASYNC_LDS
        async_cp128(ks, kdst);          // K block: verbatim row-major copy
        async_cp128(ks + 8, kdst + 8);
#else
        *(v8bf*)(kdst)     = *(const v8bf*)(ks);
        *(v8bf*)(kdst + 8) = *(const v8bf*)(ks + 8);
#endif
        const __bf16* vs = Vp + (size_t)(j * 32 + srow) * D_DIM + schunk * 16;
        v8bf v0 = *(const v8bf*)(vs), v1 = *(const v8bf*)(vs + 8);
#pragma unroll
        for (int e = 0; e < 8; ++e) {
            Vt[(schunk * 16 + e) * 40 + srow]     = v0[e];
            Vt[(schunk * 16 + 8 + e) * 40 + srow] = v1[e];
        }
#ifdef HAVE_ASYNC_LDS
        wait_async0();
#endif
        __syncthreads();

        // S = Q * K^T  (16 q-rows x 32 keys, two 16-key tiles, K-dim = d = 64)
        v8f s0 = {}, s1 = {};
#pragma unroll
        for (int c = 0; c < 2; ++c) {
            v16bf bk0 = ldfrag(&Ks[lane15 * 72 + c * 32 + hi * 16]);
            v16bf bk1 = ldfrag(&Ks[(16 + lane15) * 72 + c * 32 + hi * 16]);
            s0 = wmma_bf16(aq[c], bk0, s0);
            s1 = wmma_bf16(aq[c], bk1, s1);
        }

        // Online softmax (scores already carry D^-0.5 * log2 e).
        float p0[8], p1[8];
#pragma unroll
        for (int r = 0; r < 8; ++r) {
            const float t = red16_max(fmaxf(s0[r], s1[r]));
            const float mnew = fmaxf(mi[r], t);
            const float corr = exp2f(mi[r] - mnew);
            mi[r] = mnew;
            p0[r] = exp2f(s0[r] - mnew);
            p1[r] = exp2f(s1[r] - mnew);
            const float rs = red16_add(p0[r] + p1[r]);
            li[r] = li[r] * corr + rs;
#pragma unroll
            for (int n = 0; n < 4; ++n) acc[n][r] *= corr;
        }

        // C-layout P -> LDS -> A-layout fragment (wave-private, DScnt-ordered)
        __bf16* pw = &Pb[wave][0];
#pragma unroll
        for (int r = 0; r < 8; ++r) {
            pw[(r + 8 * hi) * 40 + lane15]      = (__bf16)p0[r];
            pw[(r + 8 * hi) * 40 + 16 + lane15] = (__bf16)p1[r];
        }
        v16bf pa = ldfrag(&pw[lane15 * 40 + hi * 16]);

        // O += P * V   (K-dim = 32 keys, 4 d-tiles of 16)
#pragma unroll
        for (int n = 0; n < 4; ++n) {
            v16bf bv = ldfrag(&Vt[(n * 16 + lane15) * 40 + hi * 16]);
            acc[n] = wmma_bf16(pa, bv, acc[n]);
        }
    }

    // Normalize and write context in [B, S, H*D] = [B, S, E] layout.
#pragma unroll
    for (int n = 0; n < 4; ++n)
#pragma unroll
        for (int r = 0; r < 8; ++r) {
            const float v = acc[n][r] / li[r];
            const int m = qblk * 64 + wave * 16 + r + 8 * hi;
            Ctx[((size_t)(b * S_DIM + m)) * E_DIM + h * D_DIM + n * 16 + lane15] =
                (__bf16)v;
        }
}

// ---------------------------------------------------------------------------
extern "C" void kernel_launch(void* const* d_in, const int* in_sizes, int n_in,
                              void* d_out, int out_size, void* d_ws, size_t ws_size,
                              hipStream_t stream) {
    const float* query = (const float*)d_in[0];   // [2, 2048, 1024]
    const float* w_qkv = (const float*)d_in[1];   // [3072, 1024]
    const float* b_qkv = (const float*)d_in[2];   // [3072]
    const float* w_out = (const float*)d_in[3];   // [1024, 1024]
    const float* b_out = (const float*)d_in[4];   // [1024]
    float* out = (float*)d_out;                   // [2, 2048, 1024] f32

    char* ws = (char*)d_ws;
    __bf16* Xb    = (__bf16*)ws; ws += (size_t)M_DIM * E_DIM * 2;        // 8 MB
    __bf16* Wqkvb = (__bf16*)ws; ws += (size_t)3 * E_DIM * E_DIM * 2;    // 6 MB
    __bf16* Woutb = (__bf16*)ws; ws += (size_t)E_DIM * E_DIM * 2;        // 2 MB
    __bf16* Qb    = (__bf16*)ws; ws += (size_t)M_DIM * E_DIM * 2;        // 8 MB
    __bf16* Kb    = (__bf16*)ws; ws += (size_t)M_DIM * E_DIM * 2;        // 8 MB
    __bf16* Vb    = (__bf16*)ws; ws += (size_t)M_DIM * E_DIM * 2;        // 8 MB
    __bf16* Ctx   = (__bf16*)ws; ws += (size_t)M_DIM * E_DIM * 2;        // 8 MB

    // 1) f32 -> bf16 converts
    mha_cvt_bf16<<<2048, 256, 0, stream>>>((const float4*)query, (ushort4*)Xb,
                                           (M_DIM * E_DIM) >> 2);
    mha_cvt_bf16<<<2048, 256, 0, stream>>>((const float4*)w_qkv, (ushort4*)Wqkvb,
                                           (3 * E_DIM * E_DIM) >> 2);
    mha_cvt_bf16<<<1024, 256, 0, stream>>>((const float4*)w_out, (ushort4*)Woutb,
                                           (E_DIM * E_DIM) >> 2);

    // 2) QKV projection: [4096,1024] x [3072,1024]^T, epilogue scatters Q/K/V
    mha_gemm_bf16<0><<<dim3(3 * E_DIM / 128, M_DIM / 128), 256, 0, stream>>>(
        Xb, Wqkvb, b_qkv, Qb, Kb, Vb, nullptr, E_DIM);

    // 3) Fused flash attention over all (b, h, q-block)
    mha_flash_attn<<<dim3(S_DIM / 64, H_DIM, B_DIM), 128, 0, stream>>>(
        Qb, Kb, Vb, Ctx);

    // 4) Output projection: [4096,1024] x [1024,1024]^T + bias -> f32 out
    mha_gemm_bf16<1><<<dim3(E_DIM / 128, M_DIM / 128), 256, 0, stream>>>(
        Ctx, Woutb, b_out, nullptr, nullptr, nullptr, out, E_DIM);
}
